// RowLSTM_67053029425626
// MI455X (gfx1250) — matrix operand order repaired and checked by reading, not compiled, not run on hardware
//
#include <hip/hip_runtime.h>
#include <hip/hip_bf16.h>

// ---------------------------------------------------------------------------
// RowLSTM for MI455X (gfx1250, wave32, WMMA + Tensor Data Mover).
//
//   B=16, Cin=128, S=64 (scan steps), W=64, NH=128, gates 4*NH=512.
//
// Phase 1: fs = masked 3x1 conv  ==> 1024 independent GEMMs (M=512,K=256,N=64)
// Phase 2: 64-step recurrence    ==> per (b,step) GEMM (M=512,K=384,N=64)
// ~43 GFLOP vs ~100MB traffic -> compute bound -> v_wmma_f32_16x16x32_bf16.
// fs gate tiles (64KB/step) are streamed into LDS with TENSOR_LOAD_TO_LDS,
// double-buffered so the TDM DMA for step s+1 overlaps the WMMAs of step s.
// ---------------------------------------------------------------------------

typedef __attribute__((ext_vector_type(16))) __bf16 v16bf;
typedef __attribute__((ext_vector_type(8)))  __bf16 v8bf;
typedef __attribute__((ext_vector_type(8)))  float  v8f;
typedef __attribute__((ext_vector_type(4)))  unsigned int v4u;
typedef __attribute__((ext_vector_type(8)))  int v8i;
typedef __attribute__((ext_vector_type(4)))  int v4i;

#define DEVINL __device__ __forceinline__

namespace {
constexpr int BN   = 16;    // batch
constexpr int CIN  = 128;   // input channels
constexpr int SS   = 64;    // rows / scan steps
constexpr int WW   = 64;    // width
constexpr int NH   = 128;   // hidden channels
constexpr int G4   = 512;   // 4*NH gate channels
constexpr int KI   = 256;   // itos GEMM K  (2 kernel rows * 128 ch)
constexpr int KS   = 384;   // stos GEMM K  (3 kernel taps * 128 ch)
constexpr int IPITCH = 264; // LDS pitch (elems) for imgT[w][k]   (528B, 16B mult)
constexpr int HPITCH = 136; // LDS pitch (elems) for hT[w+1][c]   (272B, 16B mult)

// Dynamic-LDS layout for the recurrence kernel (offsets are D#.lds_addr).
constexpr unsigned FS_BYTES = G4 * WW * sizeof(__bf16);      // 65536 per buffer
constexpr unsigned FS_OFF0  = 0;
constexpr unsigned FS_OFF1  = FS_BYTES;
constexpr unsigned HT_OFF   = 2 * FS_BYTES;                  // 131072
constexpr unsigned RECUR_LDS = HT_OFF + (WW + 2) * HPITCH * sizeof(__bf16);
}

union FragU { v16bf v; v8bf h[2]; };

// 16x16x32 bf16 WMMA, fp32 accumulate.
DEVINL v8f wmma_bf16(v16bf a, v16bf b, v8f c) {
  return __builtin_amdgcn_wmma_f32_16x16x32_bf16(
      /*neg_a=*/false, a, /*neg_b=*/false, b,
      /*c_mod=*/(short)0, c, /*reuse_a=*/false, /*reuse_b=*/false);
}

// A-matrix 16x32 bf16 fragment from row-major A[M][K] (contiguous k order, matched by the B loaders):
// lane L: row = row0+(L&15); 16 consecutive K [kb..kb+15], kb=(L>>4)*16.
DEVINL v16bf load_a_frag(const __bf16* A, int lda, int row0, int k0) {
  int lane = threadIdx.x & 31;
  const __bf16* p = A + (size_t)(row0 + (lane & 15)) * lda + k0 + ((lane >> 4) << 4);
  FragU u;
  u.h[0] = *(const v8bf*)p;        // K = kb .. kb+7
  u.h[1] = *(const v8bf*)(p + 8);  // K = kb+8 .. kb+15
  return u.v;
}

// B-matrix 32x16 bf16 fragment from *transposed* LDS tile T[n][k]:
// lane L: col n = n0+(L&15); 16 consecutive K starting at k0+(L>>4)*16.
DEVINL v16bf load_b_fragT(const __bf16* T, int pitch, int n0, int k0) {
  int lane = threadIdx.x & 31;
  const __bf16* p = T + (n0 + (lane & 15)) * pitch + k0 + ((lane >> 4) << 4);
  FragU u;
  u.h[0] = *(const v8bf*)p;
  u.h[1] = *(const v8bf*)(p + 8);
  return u.v;
}

// B fragment for the recurrence: k = kh*128 + c, B[k][w] = h[c][w+kh-1],
// stored as haloed transpose hT[w+1][c] (hT[0] and hT[65] are zero).
DEVINL v16bf load_b_hfrag(const __bf16* hTs, int n0, int k0) {
  int lane = threadIdx.x & 31;
  int kh   = k0 >> 7;                       // 0,1,2 (16-chunks never straddle)
  const __bf16* p = hTs + (n0 + (lane & 15) + kh) * HPITCH
                        + (k0 & 127) + ((lane >> 4) << 4);
  FragU u;
  u.h[0] = *(const v8bf*)p;
  u.h[1] = *(const v8bf*)(p + 8);
  return u.v;
}

DEVINL float sigmoidf(float x) { return 1.0f / (1.0f + __expf(-x)); }

// Tensor Data Mover: 1-D tile (nelem8 x 8 bytes) global -> LDS.
// D# layout per CDNA5 ISA; tracked by TENSORcnt.
// This toolchain declares the 6-arg builtin (g0, g1, g2, g3, g_extra, cpol).
DEVINL void tdm_load_1d(unsigned lds_off, const void* gptr, unsigned nelem8) {
  unsigned long long ga = (unsigned long long)gptr;
  v4u g0;
  g0[0] = 1u;                                   // count=1, user mode, no gather
  g0[1] = lds_off;                              // lds_addr (bytes)
  g0[2] = (unsigned)ga;                         // global_addr[31:0]
  g0[3] = (unsigned)((ga >> 32) & 0x1FFFFFFu)   // global_addr[56:32]
        | (2u << 30);                           // type = 2 ("image")
  v8i g1;
  g1[0] = (int)(3u << 16);                      // data_size = 8 bytes
  g1[1] = (int)((nelem8 & 0xFFFFu) << 16);      // tensor_dim0[15:0]
  g1[2] = (int)(((nelem8 >> 16) & 0xFFFFu)      // tensor_dim0[31:16]
        | (1u << 16));                          // tensor_dim1 = 1
  g1[3] = (int)((nelem8 & 0xFFFFu) << 16);      // tile_dim0 = nelem8
  g1[4] = 1;                                    // tile_dim1 = 1
  g1[5] = (int)nelem8;                          // tensor_dim0_stride[31:0]
  g1[6] = 0;
  g1[7] = 0;
  v4i z4 = {0, 0, 0, 0};
  v8i z8 = {0, 0, 0, 0, 0, 0, 0, 0};
  __builtin_amdgcn_tensor_load_to_lds(g0, g1, z4, z4, z8, 0);
}

// ---------------------------------------------------------------------------
// Kernel 1: repack conv weights (mask drops kh=2 for itos) into bf16 GEMM A.
//   A_itos[oc][kh*128+c], kh in {0,1};  A_stos[oc][kh*128+c], kh in {0,1,2}.
// ---------------------------------------------------------------------------
__global__ void prep_weights(const float* __restrict__ wi,
                             const float* __restrict__ ws,
                             __bf16* __restrict__ Ai,
                             __bf16* __restrict__ As) {
  int tid = blockIdx.x * blockDim.x + threadIdx.x;
  if (tid < G4 * KI) {
    int oc = tid >> 8, k = tid & 255;
    int kh = k >> 7,  c = k & 127;
    Ai[tid] = (__bf16)wi[(oc * CIN + c) * 3 + kh];
  }
  if (tid < G4 * KS) {
    int oc = tid / KS, k = tid - oc * KS;
    int kh = k >> 7,  c = k & 127;
    As[tid] = (__bf16)ws[(oc * NH + c) * 3 + kh];
  }
}

// ---------------------------------------------------------------------------
// Kernel 2: input-to-state conv as WMMA GEMM. One block per (b,s).
// Each wave owns a 32x64 macro-tile (2 M-tiles x 4 N-tiles): A fragments are
// reused 4x, B fragments 2x -> 12 x b128 loads per 8 WMMAs.
// ---------------------------------------------------------------------------
__global__ __launch_bounds__(512)
void itos_kernel(const float* __restrict__ image,
                 const __bf16* __restrict__ Ai,
                 const float* __restrict__ b_itos,
                 __bf16* __restrict__ fs) {
  __shared__ __bf16 imgT[WW * IPITCH];      // 33792 B

  int bs = blockIdx.x;
  int b = bs >> 6, s = bs & 63;
  int tid = threadIdx.x;
  int lane = tid & 31, wv = tid >> 5;
  int lo = lane & 15, hi = lane >> 4;

  // Warm the (L2-resident) weight matrix into near caches.
  __builtin_prefetch(Ai + (size_t)tid * 256, 0, 3);

  // Stage B matrix transposed: imgT[w][kh*128+c] = image[b][c][s-1+kh][w].
  // w is the fast index of i -> coalesced global reads.
  for (int i = tid; i < KI * WW; i += 512) {
    int k = i >> 6, w = i & 63;
    int kh = k >> 7, c = k & 127;
    int row = s - 1 + kh;
    float v = (row >= 0)
        ? image[(((size_t)b * CIN + c) * SS + row) * WW + w] : 0.0f;
    imgT[w * IPITCH + k] = (__bf16)v;
  }
  __syncthreads();

  // Wave wv -> M rows [wv*32, wv*32+32), all 4 N-tiles.
  int m0 = wv * 32;
  v8f acc[2][4];
#pragma unroll
  for (int mt = 0; mt < 2; ++mt)
#pragma unroll
    for (int nt = 0; nt < 4; ++nt)
#pragma unroll
      for (int r = 0; r < 8; ++r)
        acc[mt][nt][r] = b_itos[m0 + mt * 16 + r + hi * 8];

#pragma unroll
  for (int k0 = 0; k0 < KI; k0 += 32) {
    v16bf a0 = load_a_frag(Ai, KI, m0, k0);
    v16bf a1 = load_a_frag(Ai, KI, m0 + 16, k0);
#pragma unroll
    for (int nt = 0; nt < 4; ++nt) {
      v16bf bf = load_b_fragT(imgT, IPITCH, nt * 16, k0);
      acc[0][nt] = wmma_bf16(a0, bf, acc[0][nt]);
      acc[1][nt] = wmma_bf16(a1, bf, acc[1][nt]);
    }
  }

#pragma unroll
  for (int mt = 0; mt < 2; ++mt) {
    __bf16* fsp = fs + ((size_t)bs * G4 + m0 + mt * 16) * WW;
#pragma unroll
    for (int nt = 0; nt < 4; ++nt)
#pragma unroll
      for (int r = 0; r < 8; ++r)
        fsp[(r + hi * 8) * WW + nt * 16 + lo] = (__bf16)acc[mt][nt][r];
  }
}

// ---------------------------------------------------------------------------
// Kernel 3: 64-step recurrence. One persistent block per batch element.
// 32 waves; wave wv owns the 16x16 (ch,w) unit (ch16=wv>>2, w16=wv&3) and
// computes all four gate tiles for it. Cell state c stays in VGPRs; h lives
// in LDS as haloed transpose hT[w+1][c] (bf16), rebuilt every step. The fs
// gate tile for each step is DMA'd into a double-buffered LDS region by the
// Tensor Data Mover (wave 0 issues, TENSORcnt-gated, overlapped one step).
// ---------------------------------------------------------------------------
__global__ __launch_bounds__(1024)
void recur_kernel(const __bf16* __restrict__ As,
                  const float* __restrict__ b_stos,
                  const __bf16* __restrict__ fs,
                  const float* __restrict__ h0,
                  const float* __restrict__ c0,
                  float* __restrict__ out) {
  extern __shared__ char smem[];
  __bf16* hT = (__bf16*)(smem + HT_OFF);    // (WW+2) x HPITCH, zero halo rows

  int b = blockIdx.x;
  int tid = threadIdx.x;
  int lane = tid & 31, wv = tid >> 5;
  int lo = lane & 15, hi = lane >> 4;

  const __bf16* fs_b = fs + (size_t)b * SS * G4 * WW;

  // Kick off the TDM for step 0 while we stage h0.
  if (wv == 0)
    tdm_load_1d(FS_OFF0, fs_b, FS_BYTES / 8);

  // Stage h0 (broadcast over batch) transposed with zero halo columns.
  for (int i = tid; i < (WW + 2) * NH; i += 1024) {
    int wp = i >> 7, c = i & 127;
    float v = (wp >= 1 && wp <= WW) ? h0[c * WW + (wp - 1)] : 0.0f;
    hT[wp * HPITCH + c] = (__bf16)v;
  }

  int m_ch0 = (wv >> 2) * 16;          // hidden-channel block
  int n0    = (wv & 3) * 16;           // width block
  int w     = n0 + lo;                 // this lane's width column (C layout)

  // Persistent cell state (fp32 in VGPRs) + hoisted stos bias.
  v8f cst, bval[4];
#pragma unroll
  for (int r = 0; r < 8; ++r) {
    int ch = m_ch0 + r + hi * 8;
    cst[r] = c0[ch * WW + w];
#pragma unroll
    for (int g = 0; g < 4; ++g) bval[g][r] = b_stos[g * NH + ch];
  }

  for (int s = 0; s < SS; ++s) {
    // Gate the arrival of this step's fs tile, then overlap the next DMA.
    if (wv == 0)
      __builtin_amdgcn_s_wait_tensorcnt(0);
    __syncthreads();
    if (wv == 0 && s + 1 < SS)
      tdm_load_1d((s + 1) & 1 ? FS_OFF1 : FS_OFF0,
                  fs_b + (size_t)(s + 1) * G4 * WW, FS_BYTES / 8);

    // Active double-buffer half (computed arithmetically; a pointer table over
    // dynamic LDS folds into an unsupported static initializer).
    const __bf16* fsl =
        (const __bf16*)(smem + (((s & 1) != 0) ? FS_OFF1 : FS_OFF0));

    // acc[g] seeded with fs + bias; then GEMM over K=384 (3 width taps).
    v8f acc[4];
#pragma unroll
    for (int g = 0; g < 4; ++g)
#pragma unroll
      for (int r = 0; r < 8; ++r)
        acc[g][r] = (float)fsl[(g * NH + m_ch0 + r + hi * 8) * WW + w]
                  + bval[g][r];

#pragma unroll
    for (int k0 = 0; k0 < KS; k0 += 32) {
      v16bf bf = load_b_hfrag(hT, n0, k0);   // shared across the 4 gates
#pragma unroll
      for (int g = 0; g < 4; ++g) {
        v16bf af = load_a_frag(As, KS, g * NH + m_ch0, k0);
        acc[g] = wmma_bf16(af, bf, acc[g]);
      }
    }

    // Gate math. NOTE: h_new uses the *previous* cell state (reference quirk).
    v8f hnew;
#pragma unroll
    for (int r = 0; r < 8; ++r) {
      float xi = sigmoidf(acc[0][r]);
      float xg = tanhf(acc[1][r]);
      float xf = sigmoidf(acc[2][r]);
      float xo = sigmoidf(acc[3][r]);
      float cp = cst[r];
      cst[r]  = xf * cp + xi * xg;
      hnew[r] = cp * xo;
    }

    __syncthreads();   // all hT / fs-buffer reads for this step are done
#pragma unroll
    for (int r = 0; r < 8; ++r) {
      int ch = m_ch0 + r + hi * 8;
      float hv = hnew[r];
      hT[(w + 1) * HPITCH + ch] = (__bf16)hv;                    // next step's B
      out[(((size_t)b * NH + ch) * SS + s) * WW + w] = hv;       // [B,NH,S,W]
    }
    __syncthreads();   // writes visible before next step's reads
  }
}

// ---------------------------------------------------------------------------
// Host launcher. Workspace: fs (bf16, 64MB) + repacked bf16 weights.
// ---------------------------------------------------------------------------
extern "C" void kernel_launch(void* const* d_in, const int* in_sizes, int n_in,
                              void* d_out, int out_size, void* d_ws, size_t ws_size,
                              hipStream_t stream) {
  const float* image  = (const float*)d_in[0];
  const float* w_itos = (const float*)d_in[1];
  const float* b_itos = (const float*)d_in[2];
  const float* w_stos = (const float*)d_in[3];
  const float* b_stos = (const float*)d_in[4];
  const float* h0     = (const float*)d_in[5];
  const float* c0     = (const float*)d_in[6];
  float* out = (float*)d_out;

  char* ws = (char*)d_ws;
  const size_t fs_bytes = (size_t)BN * SS * G4 * WW * sizeof(__bf16); // 67,108,864
  const size_t ai_bytes = (size_t)G4 * KI * sizeof(__bf16);           // 262,144
  __bf16* fs = (__bf16*)ws;
  __bf16* Ai = (__bf16*)(ws + fs_bytes);
  __bf16* As = (__bf16*)(ws + fs_bytes + ai_bytes);

  prep_weights<<<(G4 * KS + 255) / 256, 256, 0, stream>>>(w_itos, w_stos, Ai, As);
  itos_kernel<<<BN * SS, 512, 0, stream>>>(image, Ai, b_itos, fs);
  recur_kernel<<<BN, 1024, RECUR_LDS, stream>>>(As, b_stos, fs, h0, c0, out);
}
